// OPTEigenAttention_80719615361036
// MI455X (gfx1250) — compile-verified, matrix operand order checked by reading, not run on hardware
//
#include <hip/hip_runtime.h>
#include <hip/hip_bf16.h>

// ---------------------------------------------------------------------------
// Problem constants (from reference)
// ---------------------------------------------------------------------------
#define BATCH   2
#define SEQ     2048
#define EMB     2048
#define NHEADS  32
#define DHEAD   32          // low-rank per-head dim (RANK/NHEADS)
#define RANK    1024        // RANK_KQ == RANK_V
#define MROWS   (BATCH*SEQ) // 4096 GEMM rows

typedef __attribute__((ext_vector_type(16))) _Float16 v16h;
typedef __attribute__((ext_vector_type(8)))  _Float16 v8h;
typedef __attribute__((ext_vector_type(8)))  float    v8f;

static __device__ __forceinline__ v8h ldg8(const _Float16* p) {
    return *reinterpret_cast<const v8h*>(p);
}
static __device__ __forceinline__ v16h cat16(v8h lo, v8h hi) {
    return __builtin_shufflevector(lo, hi, 0,1,2,3,4,5,6,7,8,9,10,11,12,13,14,15);
}
// Fragment load: uniform base pointer (stays in SGPRs) + per-lane 32-bit
// element offset -> lowers to global_load_b128 v, voff, s[base] (GVS form).
static __device__ __forceinline__ v16h ldfrag2(const _Float16* b, int off) {
    return cat16(ldg8(b + off), ldg8(b + off + 16));
}
static __device__ __forceinline__ v8f wmma_f16(v16h a, v16h b, v8f c) {
    // D = A(16x32 f16) x B(32x16 f16) + C(16x16 f32)
    return __builtin_amdgcn_wmma_f32_16x16x32_f16(false, a, false, b, (short)0, c, false, false);
}

// ---------------------------------------------------------------------------
// fp32 -> f16 conversion (one bandwidth pass; everything downstream is f16)
// ---------------------------------------------------------------------------
__global__ void cvt_f32_f16(const float* __restrict__ src,
                            _Float16* __restrict__ dst, int n) {
    int i = blockIdx.x * blockDim.x + threadIdx.x;
    int stride = gridDim.x * blockDim.x;
    for (; i < n; i += stride) dst[i] = (_Float16)src[i];
}

// ---------------------------------------------------------------------------
// Fused QKV projection: Y = X @ W^T + b  (M=4096, N=1024, K=2048)
// blockDim = 256 (8 waves): 2 waves in M x 4 waves in N.
// Wave tile 32M x 64N: 2 A-frags x 4 B-frags -> 8 WMMAs per 32-K step,
// register double-buffered; scalar-base addressing + waves_per_eu(1) to
// keep everything in VGPRs (no scratch spills).
// z = 0:Q (scaled 1/8, [head][s][d]); 1:K ([head][s][d]); 2:V^T ([head][d][s]).
// ---------------------------------------------------------------------------
__global__ void __launch_bounds__(256, 1)
gemm_qkv(const _Float16* __restrict__ Xh,
         const _Float16* __restrict__ Wqh,
         const _Float16* __restrict__ Wkh,
         const _Float16* __restrict__ Wvh,
         const float* __restrict__ bq,
         const float* __restrict__ bk,
         const float* __restrict__ bv,
         _Float16* __restrict__ Qh,
         _Float16* __restrict__ Kh,
         _Float16* __restrict__ Vt) {
    const int z = blockIdx.z;
    const _Float16* W   = (z == 0) ? Wqh : (z == 1) ? Wkh : Wvh;
    const float*    bia = (z == 0) ? bq  : (z == 1) ? bk  : bv;

    const int lane = threadIdx.x & 31;
    const int wave = threadIdx.x >> 5;
    const int col  = lane & 15;
    const int half = lane >> 4;
    const int base = 8 * half;

    const int m0 = blockIdx.y * 64 + (wave >> 2) * 32;
    const int n0 = blockIdx.x * 256 + (wave & 3) * 64;

    // per-lane constant element offsets (base folded in)
    const int offA0 = (m0 + col) * EMB + base;
    const int offA1 = offA0 + 16 * EMB;
    const int offB0 = (n0 + col) * EMB + base;
    const int offB1 = offB0 + 16 * EMB;
    const int offB2 = offB0 + 32 * EMB;
    const int offB3 = offB0 + 48 * EMB;

    v8f acc[8] = {};
    // prologue: fragments for k0 = 0
    v16h a0 = ldfrag2(Xh, offA0), a1 = ldfrag2(Xh, offA1);
    v16h b0 = ldfrag2(W, offB0), b1 = ldfrag2(W, offB1);
    v16h b2 = ldfrag2(W, offB2), b3 = ldfrag2(W, offB3);

    for (int k0 = 32; k0 < EMB; k0 += 32) {
        const _Float16* Xk = Xh + k0;   // uniform: SALU increment only
        const _Float16* Wk = W + k0;
        v16h na0 = ldfrag2(Xk, offA0), na1 = ldfrag2(Xk, offA1);
        v16h nb0 = ldfrag2(Wk, offB0), nb1 = ldfrag2(Wk, offB1);
        v16h nb2 = ldfrag2(Wk, offB2), nb3 = ldfrag2(Wk, offB3);

        acc[0] = wmma_f16(a0, b0, acc[0]);
        acc[1] = wmma_f16(a1, b0, acc[1]);
        acc[2] = wmma_f16(a0, b1, acc[2]);
        acc[3] = wmma_f16(a1, b1, acc[3]);
        acc[4] = wmma_f16(a0, b2, acc[4]);
        acc[5] = wmma_f16(a1, b2, acc[5]);
        acc[6] = wmma_f16(a0, b3, acc[6]);
        acc[7] = wmma_f16(a1, b3, acc[7]);

        a0 = na0; a1 = na1; b0 = nb0; b1 = nb1; b2 = nb2; b3 = nb3;
    }
    acc[0] = wmma_f16(a0, b0, acc[0]);
    acc[1] = wmma_f16(a1, b0, acc[1]);
    acc[2] = wmma_f16(a0, b1, acc[2]);
    acc[3] = wmma_f16(a1, b1, acc[3]);
    acc[4] = wmma_f16(a0, b2, acc[4]);
    acc[5] = wmma_f16(a1, b2, acc[5]);
    acc[6] = wmma_f16(a0, b3, acc[6]);
    acc[7] = wmma_f16(a1, b3, acc[7]);

    const float scale = (z == 0) ? 0.125f : 1.0f;  // HEAD_DIM(64)^-0.5 on Q
#pragma unroll
    for (int t = 0; t < 4; ++t) {
        const int n = n0 + 16 * t + col;
        const float bb = bia[n];
        const int h = n >> 5, d = n & 31;
#pragma unroll
        for (int u = 0; u < 2; ++u) {
#pragma unroll
            for (int r = 0; r < 8; ++r) {
                const int m = m0 + 16 * u + r + 8 * half;
                const int b_ = m >> 11;          // m / SEQ
                const int s  = m & (SEQ - 1);
                const int head = b_ * NHEADS + h;
                const _Float16 hv = (_Float16)((acc[t * 2 + u][r] + bb) * scale);
                if (z == 2)
                    Vt[((size_t)head * DHEAD + d) * SEQ + s] = hv;
                else if (z == 0)
                    Qh[((size_t)head * SEQ + s) * DHEAD + d] = hv;
                else
                    Kh[((size_t)head * SEQ + s) * DHEAD + d] = hv;
            }
        }
    }
}

// ---------------------------------------------------------------------------
// Causal flash attention per head. One wave owns one 16-query tile,
// 4 waves per block. Per 32-key step: K and V fragment loads are issued
// up front (V overlaps score WMMAs + softmax VALU), then 2 score WMMAs,
// online softmax via 16-lane shuffles, P transposed through wave-private
// LDS (in-order per wave, no barrier), then 2 P*V WMMAs.
// ---------------------------------------------------------------------------
__global__ void attn_fwd(const _Float16* __restrict__ Qh,
                         const _Float16* __restrict__ Kh,
                         const _Float16* __restrict__ Vt,
                         _Float16* __restrict__ AO) {
    __shared__ __align__(16) _Float16 Plds[4][16][32];

    const int lane = threadIdx.x & 31;
    const int wave = threadIdx.x >> 5;
    const int col  = lane & 15;
    const int half = lane >> 4;
    const int base = 8 * half;

    const int q0   = (blockIdx.x * 4 + wave) * 16;
    const int head = blockIdx.y;                       // 0..B*H-1
    const _Float16* Qp = Qh + (size_t)head * SEQ * DHEAD;
    const _Float16* Kp = Kh + (size_t)head * SEQ * DHEAD;
    const _Float16* Vp = Vt + (size_t)head * DHEAD * SEQ;

    // Q fragment (rows q0..q0+15, K-dim = DHEAD = 32): resident all loop long
    const v16h qa = ldfrag2(Qp, (q0 + col) * DHEAD + base);

    // constant per-lane element offsets
    const int koff0 = col * DHEAD + base;          // + j0*DHEAD (uniform)
    const int koff1 = koff0 + 16 * DHEAD;
    const int voff0 = col * SEQ + base;            // + j0 (uniform)
    const int voff1 = voff0 + 16 * SEQ;

    float mrow[8], lrow[8];
#pragma unroll
    for (int r = 0; r < 8; ++r) { mrow[r] = -1.0e30f; lrow[r] = 0.0f; }
    v8f o0 = {}, o1 = {};
    const v8f czero = {};

    for (int j0 = 0; j0 <= q0 + 15; j0 += 32) {
        const _Float16* Kj = Kp + (size_t)j0 * DHEAD;   // uniform bases
        const _Float16* Vj = Vp + j0;

        // --- issue all global loads for this step up front ---
        v16h kb0 = ldfrag2(Kj, koff0);
        v16h kb1 = ldfrag2(Kj, koff1);
        v16h vb0 = ldfrag2(Vj, voff0);   // overlaps score WMMAs + softmax
        v16h vb1 = ldfrag2(Vj, voff1);

        // --- scores: Q (16x32) x K^T -> two 16x16 tiles ---
        v8f s0 = wmma_f16(qa, kb0, czero);
        v8f s1 = wmma_f16(qa, kb1, czero);

        // --- causal mask (only partial tiles need it) ---
        if (j0 + 31 > q0) {
#pragma unroll
            for (int r = 0; r < 8; ++r) {
                const int rq = q0 + r + 8 * half;
                if (j0 + col      > rq) s0[r] = -1.0e30f;
                if (j0 + 16 + col > rq) s1[r] = -1.0e30f;
            }
        }

        // --- online softmax update (rows live in lane halves) ---
        float p0[8], p1[8];
#pragma unroll
        for (int r = 0; r < 8; ++r) {
            float t = fmaxf(s0[r], s1[r]);
            t = fmaxf(t, __shfl_xor(t, 1));
            t = fmaxf(t, __shfl_xor(t, 2));
            t = fmaxf(t, __shfl_xor(t, 4));
            t = fmaxf(t, __shfl_xor(t, 8));
            const float mn = fmaxf(mrow[r], t);
            const float alpha = __expf(mrow[r] - mn);
            p0[r] = __expf(s0[r] - mn);
            p1[r] = __expf(s1[r] - mn);
            float rs = p0[r] + p1[r];
            rs += __shfl_xor(rs, 1);
            rs += __shfl_xor(rs, 2);
            rs += __shfl_xor(rs, 4);
            rs += __shfl_xor(rs, 8);
            lrow[r] = lrow[r] * alpha + rs;
            mrow[r] = mn;
            o0[r] *= alpha;
            o1[r] *= alpha;
        }

        // --- transpose P through wave-private LDS (in-order per wave) ---
#pragma unroll
        for (int r = 0; r < 8; ++r) {
            const int rm = r + 8 * half;
            Plds[wave][rm][col]      = (_Float16)p0[r];
            Plds[wave][rm][col + 16] = (_Float16)p1[r];
        }
        const _Float16* pp = &Plds[wave][col][0];
        v16h pa = cat16(*reinterpret_cast<const v8h*>(pp + base),
                        *reinterpret_cast<const v8h*>(pp + base + 16));

        // --- O += P (16x32 keys) x V (32 keys x 32 dv) ---
        o0 = wmma_f16(pa, vb0, o0);
        o1 = wmma_f16(pa, vb1, o1);
    }

    // --- epilogue: normalize and emit f16 in (b, s, h*32+d) layout ---
    const int b_ = head >> 5, h = head & 31;
#pragma unroll
    for (int r = 0; r < 8; ++r) {
        const int s = q0 + r + 8 * half;
        const float inv = 1.0f / lrow[r];
        _Float16* dst = AO + ((size_t)(b_ * SEQ + s)) * RANK + h * DHEAD;
        dst[col]      = (_Float16)(o0[r] * inv);
        dst[col + 16] = (_Float16)(o1[r] * inv);
    }
}

// ---------------------------------------------------------------------------
// Output projection: Y = AO @ Wo^T + bo (M=4096, N=2048, K=1024), fp32 out.
// Same 32M x 64N wave tile + double buffering + scalar-base addressing.
// ---------------------------------------------------------------------------
__global__ void __launch_bounds__(256, 1)
gemm_out(const _Float16* __restrict__ AO,
         const _Float16* __restrict__ Woh,
         const float* __restrict__ bo,
         float* __restrict__ Y) {
    const int lane = threadIdx.x & 31;
    const int wave = threadIdx.x >> 5;
    const int col  = lane & 15;
    const int half = lane >> 4;
    const int base = 8 * half;

    const int m0 = blockIdx.y * 64 + (wave >> 2) * 32;
    const int n0 = blockIdx.x * 256 + (wave & 3) * 64;

    const int offA0 = (m0 + col) * RANK + base;
    const int offA1 = offA0 + 16 * RANK;
    const int offB0 = (n0 + col) * RANK + base;
    const int offB1 = offB0 + 16 * RANK;
    const int offB2 = offB0 + 32 * RANK;
    const int offB3 = offB0 + 48 * RANK;

    v8f acc[8] = {};
    v16h a0 = ldfrag2(AO, offA0), a1 = ldfrag2(AO, offA1);
    v16h b0 = ldfrag2(Woh, offB0), b1 = ldfrag2(Woh, offB1);
    v16h b2 = ldfrag2(Woh, offB2), b3 = ldfrag2(Woh, offB3);

    for (int k0 = 32; k0 < RANK; k0 += 32) {
        const _Float16* Ak = AO + k0;
        const _Float16* Wk = Woh + k0;
        v16h na0 = ldfrag2(Ak, offA0), na1 = ldfrag2(Ak, offA1);
        v16h nb0 = ldfrag2(Wk, offB0), nb1 = ldfrag2(Wk, offB1);
        v16h nb2 = ldfrag2(Wk, offB2), nb3 = ldfrag2(Wk, offB3);

        acc[0] = wmma_f16(a0, b0, acc[0]);
        acc[1] = wmma_f16(a1, b0, acc[1]);
        acc[2] = wmma_f16(a0, b1, acc[2]);
        acc[3] = wmma_f16(a1, b1, acc[3]);
        acc[4] = wmma_f16(a0, b2, acc[4]);
        acc[5] = wmma_f16(a1, b2, acc[5]);
        acc[6] = wmma_f16(a0, b3, acc[6]);
        acc[7] = wmma_f16(a1, b3, acc[7]);

        a0 = na0; a1 = na1; b0 = nb0; b1 = nb1; b2 = nb2; b3 = nb3;
    }
    acc[0] = wmma_f16(a0, b0, acc[0]);
    acc[1] = wmma_f16(a1, b0, acc[1]);
    acc[2] = wmma_f16(a0, b1, acc[2]);
    acc[3] = wmma_f16(a1, b1, acc[3]);
    acc[4] = wmma_f16(a0, b2, acc[4]);
    acc[5] = wmma_f16(a1, b2, acc[5]);
    acc[6] = wmma_f16(a0, b3, acc[6]);
    acc[7] = wmma_f16(a1, b3, acc[7]);

#pragma unroll
    for (int t = 0; t < 4; ++t) {
        const int n = n0 + 16 * t + col;
        const float bb = bo[n];
#pragma unroll
        for (int u = 0; u < 2; ++u) {
#pragma unroll
            for (int r = 0; r < 8; ++r) {
                const int m = m0 + 16 * u + r + 8 * half;
                Y[(size_t)m * EMB + n] = acc[t * 2 + u][r] + bb;
            }
        }
    }
}

// ---------------------------------------------------------------------------
// Launch
// ---------------------------------------------------------------------------
extern "C" void kernel_launch(void* const* d_in, const int* in_sizes, int n_in,
                              void* d_out, int out_size, void* d_ws, size_t ws_size,
                              hipStream_t stream) {
    const float* X  = (const float*)d_in[0];
    // d_in[1] = attention_mask (causal, applied analytically)
    const float* Wq = (const float*)d_in[2];
    const float* bq = (const float*)d_in[3];
    const float* Wk = (const float*)d_in[4];
    const float* bk = (const float*)d_in[5];
    const float* Wv = (const float*)d_in[6];
    const float* bv = (const float*)d_in[7];
    const float* Wo = (const float*)d_in[8];
    const float* bo = (const float*)d_in[9];
    float* Y = (float*)d_out;

    // Workspace carve-up (f16 elements); total = 64 MiB
    _Float16* p = (_Float16*)d_ws;
    const size_t nX  = (size_t)MROWS * EMB;                   // 8,388,608
    const size_t nW  = (size_t)RANK * EMB;                    // 2,097,152
    const size_t nHd = (size_t)BATCH * NHEADS * SEQ * DHEAD;  // 4,194,304
    _Float16* Xh  = p; p += nX;
    _Float16* Wqh = p; p += nW;
    _Float16* Wkh = p; p += nW;
    _Float16* Wvh = p; p += nW;
    _Float16* Woh = p; p += nW;
    _Float16* Qh  = p; p += nHd;
    _Float16* Kh  = p; p += nHd;
    _Float16* Vt  = p; p += nHd;
    _Float16* AO  = p; p += nHd;

    // 1) fp32 -> f16 conversions
    cvt_f32_f16<<<2048, 256, 0, stream>>>(X,  Xh,  (int)nX);
    cvt_f32_f16<<<1024, 256, 0, stream>>>(Wq, Wqh, (int)nW);
    cvt_f32_f16<<<1024, 256, 0, stream>>>(Wk, Wkh, (int)nW);
    cvt_f32_f16<<<1024, 256, 0, stream>>>(Wv, Wvh, (int)nW);
    cvt_f32_f16<<<1024, 256, 0, stream>>>(Wo, Woh, (int)nW);

    // 2) QKV projections (z selects Q/K/V)
    gemm_qkv<<<dim3(RANK / 256, MROWS / 64, 3), 256, 0, stream>>>(
        Xh, Wqh, Wkh, Wvh, bq, bk, bv, Qh, Kh, Vt);

    // 3) causal flash attention: 32 query-tile blocks x 64 heads
    attn_fwd<<<dim3(SEQ / 64, BATCH * NHEADS), 128, 0, stream>>>(Qh, Kh, Vt, AO);

    // 4) output projection
    gemm_out<<<dim3(EMB / 256, MROWS / 64), 256, 0, stream>>>(AO, Woh, bo, Y);
}
